// CrossTransformerBlock3D_29343216566564
// MI455X (gfx1250) — matrix operand order repaired and checked by reading, not compiled, not run on hardware
//
#include <hip/hip_runtime.h>
#include <hip/hip_bf16.h>

typedef _Float16 h16;
typedef __attribute__((ext_vector_type(16))) _Float16 v16h;
typedef __attribute__((ext_vector_type(8)))  _Float16 v8h;
typedef __attribute__((ext_vector_type(8)))  float    v8f;

#define EMBED 192
#define NTOK  131072          // 32*64*64 tokens
#define SCALE_Q 0.17677669529663687f

// ---------------- WMMA fragment helpers (gfx1250 wave32 layouts) ----------------

// A 16x32 f16: lanes 0-15 hold row M=lane, K {0..15}; lanes 16-31 K {16..31}
__device__ __forceinline__ v16h load_afrag(const h16* base, int row0, int ld, int k0) {
  int lane = threadIdx.x & 31;
  const h16* r = base + (size_t)(row0 + (lane & 15)) * ld + k0 + ((lane >> 4) << 4);
  return *(const v16h*)r;
}

// B 32x16 f16 from transposed (N-major) storage: lane holds column N=lane&15,
// 16 contiguous K starting at (lane>>4)*16 -> one contiguous 32B load
__device__ __forceinline__ v16h load_bfrag(const h16* baseT, int n0, int ld, int k0) {
  int lane = threadIdx.x & 31;
  const h16* r = baseT + (size_t)(n0 + (lane & 15)) * ld + k0 + ((lane >> 4) << 4);
  return *(const v16h*)r;
}

__device__ __forceinline__ v8f wmma_step(v16h a, v16h b, v8f c) {
  return __builtin_amdgcn_wmma_f32_16x16x32_f16(false, a, false, b, (short)0, c, false, false);
}

// ---------------- weight / bias prep: f16 transposed weights + RPB table ----------------

__global__ void prep_kernel(const float* __restrict__ qkv_w, const float* __restrict__ proj_w,
                            const float* __restrict__ fc1_w, const float* __restrict__ fc2_w,
                            const float* __restrict__ rpb,
                            h16* wqT, h16* wkT, h16* wvT, h16* pjT, h16* f1T, h16* f2T,
                            float* bias) {
  int id = blockIdx.x * blockDim.x + threadIdx.x;
  if (id < 192 * 192) {
    int o = id / 192, i = id % 192;
    wqT[o * 192 + i] = (h16)qkv_w[i * 576 + o];
    wkT[o * 192 + i] = (h16)qkv_w[i * 576 + 192 + o];
    wvT[o * 192 + i] = (h16)qkv_w[i * 576 + 384 + o];
    pjT[o * 192 + i] = (h16)proj_w[i * 192 + o];
  }
  if (id < 768 * 192) {
    int o = id / 192, i = id % 192;
    f1T[o * 192 + i] = (h16)fc1_w[i * 768 + o];
    int o2 = id / 768, i2 = id % 768;
    f2T[o2 * 768 + i2] = (h16)fc2_w[i2 * 192 + o2];
  }
  if (id < 6 * 64 * 64) {
    int hd = id / 4096, r = id % 4096, i = r / 64, j = r % 64;
    int dd = (i >> 4) - (j >> 4) + 3;
    int hh = ((i >> 2) & 3) - ((j >> 2) & 3) + 3;
    int ww = (i & 3) - (j & 3) + 3;
    bias[id] = rpb[(dd * 49 + hh * 7 + ww) * 6 + hd];
  }
}

// ---------------- layernorm: one wave per token, optional window-order scatter ----------------

__global__ void ln_kernel(const float* __restrict__ src, const float* __restrict__ g,
                          const float* __restrict__ b, h16* __restrict__ dst, int window_order) {
  int tok = blockIdx.x * (blockDim.x >> 5) + (threadIdx.x >> 5);
  if (tok >= NTOK) return;
  int lane = threadIdx.x & 31;
  const float* row = src + (size_t)tok * EMBED;
  float v[6], s = 0.f, s2 = 0.f;
#pragma unroll
  for (int i = 0; i < 6; ++i) { float x = row[lane + 32 * i]; v[i] = x; s += x; s2 += x * x; }
#pragma unroll
  for (int off = 16; off > 0; off >>= 1) {
    s  += __shfl_xor(s,  off, 32);
    s2 += __shfl_xor(s2, off, 32);
  }
  float m = s * (1.f / 192.f);
  float inv = rsqrtf(s2 * (1.f / 192.f) - m * m + 1e-5f);
  size_t dtok = tok;
  if (window_order) {
    int d = tok >> 12, hh = (tok >> 6) & 63, w = tok & 63;
    int win = ((d >> 2) * 16 + (hh >> 2)) * 16 + (w >> 2);
    int r = ((d & 3) << 4) + ((hh & 3) << 2) + (w & 3);
    dtok = (size_t)win * 64 + r;
  }
  h16* drow = dst + dtok * EMBED;
#pragma unroll
  for (int i = 0; i < 6; ++i) {
    int c = lane + 32 * i;
    drow[c] = (h16)((v[i] - m) * inv * g[c] + b[c]);
  }
}

// ---------------- fused QKV + attention, one wave per (window, head) ----------------

__global__ void __launch_bounds__(32) attn_kernel(const h16* __restrict__ xn, const h16* __restrict__ yn,
                                                  const h16* __restrict__ wqT, const h16* __restrict__ wkT,
                                                  const h16* __restrict__ wvT, const float* __restrict__ qkv_b,
                                                  const float* __restrict__ bias, h16* __restrict__ attn_out) {
  int win  = blockIdx.x / 6;
  int head = blockIdx.x % 6;
  int lane = threadIdx.x & 31;
  int l15 = lane & 15;
  int hi8 = (lane >> 4) << 3;

  __shared__ float smem_f[7168];            // 28 KB
  h16*   qh = (h16*)smem_f;                 // [64][32]
  h16*   kh = qh + 2048;                    // [64][32]
  h16*   vT = kh + 2048;                    // [32][64]  (V transposed -> B^T for attn*V)
  float* sc = smem_f + 3072;                // [64][64] f32 scores
  h16*   pr = (h16*)smem_f;                 // [64][64] f16 probs, overlays qh+kh (dead by then)

  const h16* xw = xn + (size_t)win * 64 * EMBED;
  const h16* yw = yn + (size_t)win * 64 * EMBED;
  int cb = head * 32;

  // ---- q = LN(y)*Wq (scaled), k = LN(x)*Wk, vT = (LN(x)*Wv)^T ----
  for (int m = 0; m < 4; ++m)
    for (int n = 0; n < 2; ++n) {
      v8f aq = {}, ak = {}, av = {};
      for (int kk = 0; kk < 6; ++kk) {
        v16h ax = load_afrag(xw, 16 * m, EMBED, 32 * kk);
        v16h ay = load_afrag(yw, 16 * m, EMBED, 32 * kk);
        v16h bq = load_bfrag(wqT, cb + 16 * n, EMBED, 32 * kk);
        v16h bk = load_bfrag(wkT, cb + 16 * n, EMBED, 32 * kk);
        v16h bv = load_bfrag(wvT, cb + 16 * n, EMBED, 32 * kk);
        aq = wmma_step(ay, bq, aq);
        ak = wmma_step(ax, bk, ak);
        av = wmma_step(ax, bv, av);
      }
      float qb = qkv_b[cb + 16 * n + l15];
      float kb = qkv_b[192 + cb + 16 * n + l15];
      float vb = qkv_b[384 + cb + 16 * n + l15];
#pragma unroll
      for (int r = 0; r < 8; ++r) {
        int row = 16 * m + r + hi8;
        qh[row * 32 + 16 * n + l15] = (h16)((aq[r] + qb) * SCALE_Q);
        kh[row * 32 + 16 * n + l15] = (h16)(ak[r] + kb);
        vT[(16 * n + l15) * 64 + row] = (h16)(av[r] + vb);   // free transpose on D-store
      }
    }

  // ---- scores = q @ k^T + rel-pos bias ----
  const float* bh = bias + head * 4096;
  for (int ma = 0; ma < 4; ++ma)
    for (int na = 0; na < 4; ++na) {
      v8f acc = {};
      v16h a = load_afrag(qh, 16 * ma, 32, 0);
      v16h b = load_bfrag(kh, 16 * na, 32, 0);   // k row-major == B^T for q@k^T
      acc = wmma_step(a, b, acc);
#pragma unroll
      for (int r = 0; r < 8; ++r) {
        int row = 16 * ma + r + hi8, col = 16 * na + l15;
        sc[row * 64 + col] = acc[r] + bh[row * 64 + col];
      }
    }

  // ---- row softmax (2 rows per lane) ----
  for (int row = lane; row < 64; row += 32) {
    float mx = -1e30f;
    for (int c = 0; c < 64; ++c) mx = fmaxf(mx, sc[row * 64 + c]);
    float sum = 0.f;
    for (int c = 0; c < 64; ++c) { float e = __expf(sc[row * 64 + c] - mx); sum += e; sc[row * 64 + c] = e; }
    float rinv = 1.f / sum;
    for (int c = 0; c < 64; ++c) pr[row * 64 + c] = (h16)(sc[row * 64 + c] * rinv);
  }

  // ---- out = probs @ v ----
  for (int m = 0; m < 4; ++m)
    for (int n = 0; n < 2; ++n) {
      v8f acc = {};
      for (int kk = 0; kk < 2; ++kk) {
        v16h a = load_afrag(pr, 16 * m, 64, 32 * kk);
        v16h b = load_bfrag(vT, 16 * n, 64, 32 * kk);
        acc = wmma_step(a, b, acc);
      }
#pragma unroll
      for (int r = 0; r < 8; ++r) {
        int row = 16 * m + r + hi8;
        attn_out[((size_t)win * 64 + row) * EMBED + cb + 16 * n + l15] = (h16)acc[r];
      }
    }
}

// ---------------- proj + window_reverse scatter + residual -> x1 (in d_out) ----------------

__global__ void __launch_bounds__(128) proj_kernel(const h16* __restrict__ ao, const h16* __restrict__ pjT,
                                                   const float* __restrict__ proj_b,
                                                   const float* __restrict__ x_in, float* __restrict__ out) {
  int win = blockIdx.x;
  int wv = threadIdx.x >> 5;
  int lane = threadIdx.x & 31;
  int l15 = lane & 15, hi8 = (lane >> 4) << 3;
  __shared__ int soff[64];
  if (threadIdx.x < 64) {
    int r = threadIdx.x;
    int d = (win >> 8) * 4 + (r >> 4);
    int h = ((win >> 4) & 15) * 4 + ((r >> 2) & 3);
    int w = (win & 15) * 4 + (r & 3);
    soff[r] = (d * 64 + h) * 64 + w;
  }
  __syncthreads();
  const h16* arow = ao + (size_t)win * 64 * EMBED;
  v8f acc[12];
  for (int n = 0; n < 12; ++n) acc[n] = (v8f){};
  for (int kk = 0; kk < 6; ++kk) {
    v16h a = load_afrag(arow, 16 * wv, EMBED, 32 * kk);   // reuse A across 12 N tiles
    for (int n = 0; n < 12; ++n) {
      v16h b = load_bfrag(pjT, 16 * n, EMBED, 32 * kk);
      acc[n] = wmma_step(a, b, acc[n]);
    }
  }
  for (int n = 0; n < 12; ++n) {
    int c = 16 * n + l15;
    float pb = proj_b[c];
#pragma unroll
    for (int r = 0; r < 8; ++r) {
      size_t so = (size_t)soff[16 * wv + r + hi8] * EMBED + c;
      out[so] = x_in[so] + acc[n][r] + pb;
    }
  }
}

// ---------------- fused MLP: fc1 + GELU (LDS-staged) + fc2 + residual ----------------

__global__ void __launch_bounds__(128) mlp_kernel(const h16* __restrict__ xn2, const h16* __restrict__ f1T,
                                                  const float* __restrict__ fc1_b, const h16* __restrict__ f2T,
                                                  const float* __restrict__ fc2_b, float* __restrict__ out) {
  int tile = blockIdx.x;                    // 64-token tile (spatial order)
  int wv = threadIdx.x >> 5;
  int lane = threadIdx.x & 31;
  int l15 = lane & 15, hi8 = (lane >> 4) << 3;
  __shared__ h16 hbuf[64 * EMBED];          // 24 KB hidden chunk
  const h16* xrow = xn2 + (size_t)tile * 64 * EMBED;
  v8f acc[12];
  for (int n = 0; n < 12; ++n) acc[n] = (v8f){};
  for (int ch = 0; ch < 4; ++ch) {          // 4 chunks of 192 hidden channels
    for (int n = 0; n < 12; ++n) {
      v8f hacc = {};
      for (int kk = 0; kk < 6; ++kk) {
        v16h a = load_afrag(xrow, 16 * wv, EMBED, 32 * kk);
        v16h b = load_bfrag(f1T, ch * 192 + 16 * n, EMBED, 32 * kk);
        hacc = wmma_step(a, b, hacc);
      }
      float fb = fc1_b[ch * 192 + 16 * n + l15];
#pragma unroll
      for (int r = 0; r < 8; ++r) {
        float x = hacc[r] + fb;
        float gel = 0.5f * x * (1.f + erff(x * 0.70710678118f));   // exact GELU
        hbuf[(16 * wv + r + hi8) * EMBED + 16 * n + l15] = (h16)gel;
      }
    }
    __syncthreads();
    for (int kk = 0; kk < 6; ++kk) {
      v16h a = load_afrag(hbuf, 16 * wv, EMBED, 32 * kk);
      for (int n = 0; n < 12; ++n) {
        v16h b = load_bfrag(f2T, 16 * n, 768, ch * 192 + 32 * kk);
        acc[n] = wmma_step(a, b, acc[n]);
      }
    }
    __syncthreads();
  }
  for (int n = 0; n < 12; ++n) {
    int c = 16 * n + l15;
    float fb = fc2_b[c];
#pragma unroll
    for (int r = 0; r < 8; ++r) {
      size_t so = ((size_t)tile * 64 + 16 * wv + r + hi8) * EMBED + c;
      out[so] = out[so] + acc[n][r] + fb;    // out currently holds x1 (written by proj this call)
    }
  }
}

// ---------------- launch ----------------

extern "C" void kernel_launch(void* const* d_in, const int* in_sizes, int n_in,
                              void* d_out, int out_size, void* d_ws, size_t ws_size,
                              hipStream_t stream) {
  (void)in_sizes; (void)n_in; (void)out_size; (void)ws_size;
  const float* x      = (const float*)d_in[0];
  const float* y      = (const float*)d_in[1];
  const float* n1g    = (const float*)d_in[3];
  const float* n1b    = (const float*)d_in[4];
  const float* qkv_w  = (const float*)d_in[5];
  const float* qkv_b  = (const float*)d_in[6];
  const float* rpb    = (const float*)d_in[7];
  const float* proj_w = (const float*)d_in[8];
  const float* proj_b = (const float*)d_in[9];
  const float* n2g    = (const float*)d_in[10];
  const float* n2b    = (const float*)d_in[11];
  const float* fc1_w  = (const float*)d_in[12];
  const float* fc1_b  = (const float*)d_in[13];
  const float* fc2_w  = (const float*)d_in[14];
  const float* fc2_b  = (const float*)d_in[15];
  float* out = (float*)d_out;
  char*  ws  = (char*)d_ws;

  const size_t ACT = (size_t)NTOK * EMBED * sizeof(h16);   // 50331648
  h16*   xn   = (h16*)(ws);
  h16*   yn   = (h16*)(ws + ACT);
  h16*   ao   = (h16*)(ws + 2 * ACT);
  h16*   xn2  = (h16*)(ws + 3 * ACT);
  size_t woff = 4 * ACT;
  h16*   wqT  = (h16*)(ws + woff);                 woff += 192 * 192 * 2;
  h16*   wkT  = (h16*)(ws + woff);                 woff += 192 * 192 * 2;
  h16*   wvT  = (h16*)(ws + woff);                 woff += 192 * 192 * 2;
  h16*   pjT  = (h16*)(ws + woff);                 woff += 192 * 192 * 2;
  h16*   f1T  = (h16*)(ws + woff);                 woff += 768 * 192 * 2;
  h16*   f2T  = (h16*)(ws + woff);                 woff += 192 * 768 * 2;
  float* bias = (float*)(ws + woff);

  prep_kernel<<<576, 256, 0, stream>>>(qkv_w, proj_w, fc1_w, fc2_w, rpb,
                                       wqT, wkT, wvT, pjT, f1T, f2T, bias);
  ln_kernel<<<NTOK / 8, 256, 0, stream>>>(x, n1g, n1b, xn, 1);
  ln_kernel<<<NTOK / 8, 256, 0, stream>>>(y, n1g, n1b, yn, 1);
  attn_kernel<<<2048 * 6, 32, 0, stream>>>(xn, yn, wqT, wkT, wvT, qkv_b, bias, ao);
  proj_kernel<<<2048, 128, 0, stream>>>(ao, pjT, proj_b, x, out);
  ln_kernel<<<NTOK / 8, 256, 0, stream>>>(out, n2g, n2b, xn2, 0);
  mlp_kernel<<<2048, 128, 0, stream>>>(xn2, f1T, fc1_b, f2T, fc2_b, out);
}
